// GroupedQueryAttention_55224689492074
// MI455X (gfx1250) — matrix operand order repaired and checked by reading, not compiled, not run on hardware
//
#include <hip/hip_runtime.h>
#include <hip/hip_bf16.h>
#include <stdint.h>

#define B_   2
#define L_   2048
#define D_   2048
#define H_   16
#define G_   4
#define HD_  128
#define EPS_ 1e-6f
#define SM_SCALE (1.0f / 16384.0f)   // 1 / HD^2

typedef __bf16 bf16;
typedef __attribute__((ext_vector_type(16))) __bf16 v16bf;
typedef __attribute__((ext_vector_type(8)))  __bf16 bf16x8;
typedef __attribute__((ext_vector_type(8)))  float  v8f;
typedef __attribute__((ext_vector_type(4)))  unsigned int u32x4;
typedef __attribute__((ext_vector_type(8)))  int i32x8;
typedef __attribute__((ext_vector_type(4)))  int i32x4;

#if defined(__gfx1250__) && __has_builtin(__builtin_amdgcn_tensor_load_to_lds) && \
    __has_builtin(__builtin_amdgcn_s_wait_tensorcnt)
#define HAVE_TDM 1
#else
#define HAVE_TDM 0
#endif

__device__ __forceinline__ bf16 f2bf(float f) {
  unsigned u = __builtin_bit_cast(unsigned, f);
  unsigned r = u + 0x7FFFu + ((u >> 16) & 1u);      // round-to-nearest-even
  unsigned short s = (unsigned short)(r >> 16);
  return __builtin_bit_cast(bf16, s);
}

__device__ __forceinline__ v8f v8f_zero() {
  v8f z;
#pragma unroll
  for (int i = 0; i < 8; ++i) z[i] = 0.0f;
  return z;
}

__device__ __forceinline__ v16bf make_frag(bf16x8 lo, bf16x8 hi) {
  v16bf r;
#pragma unroll
  for (int i = 0; i < 8; ++i) { r[i] = lo[i]; r[i + 8] = hi[i]; }
  return r;
}

// A-matrix 16x32 bf16 fragment (ISA 7.12.2): lane holds row (lane&15);
// elements 0..7 = K[half*8 .. +8), elements 8..15 = K[16+half*8 .. +8).
__device__ __forceinline__ v16bf frag_a_ld(const bf16* rowk, int half) {
  bf16x8 lo = *(const bf16x8*)(rowk + half * 8);
  bf16x8 hi = *(const bf16x8*)(rowk + 16 + half * 8);
  return make_frag(lo, hi);
}

// B-matrix 32x16 bf16 fragment from [N][K]-major storage: lane holds column
// (lane&15); elements 0..7 = K[half*8 .. +8), elements 8..15 = K[16+half*8 .. +8).
__device__ __forceinline__ v16bf frag_b_ld(const bf16* coln, int half) {
  bf16x8 lo = *(const bf16x8*)(coln + half * 8);
  bf16x8 hi = *(const bf16x8*)(coln + 16 + half * 8);
  return make_frag(lo, hi);
}

#if HAVE_TDM
// Issue a TDM 2D tile load (bf16 elements) global -> LDS. One wave only.
// D# group0: count=1, lds_addr, 57-bit global byte address, type=2 ("image").
// D# group1: data_size=2B, tensor dims == tile dims, dim0 stride in elements.
__device__ __forceinline__ void tdm_load_2d(unsigned lds_addr, unsigned long long ga,
                                            unsigned tw, unsigned th,
                                            unsigned long long stride_elems)
{
  u32x4 g0;
  g0[0] = 1u;                                            // count=1 (user D#)
  g0[1] = lds_addr;                                      // [63:32] lds_addr
  g0[2] = (unsigned)ga;                                  // [95:64] gaddr lo
  g0[3] = ((unsigned)(ga >> 32) & 0x01FFFFFFu) | 0x80000000u; // gaddr hi | type=2
  i32x8 g1;
  g1[0] = (int)(1u << 16);                               // data_size=1 (2 bytes)
  g1[1] = (int)((tw & 0xFFFFu) << 16);                   // tensor_dim0 lo16
  g1[2] = (int)(((tw >> 16) & 0xFFFFu) | ((th & 0xFFFFu) << 16)); // td0 hi | td1 lo
  g1[3] = (int)(((th >> 16) & 0xFFFFu) | ((tw & 0xFFFFu) << 16)); // td1 hi | tile_dim0
  g1[4] = (int)(th & 0xFFFFu);                           // tile_dim1 | tile_dim2=0
  g1[5] = (int)(unsigned)stride_elems;                   // tensor_dim0_stride lo32
  g1[6] = (int)((unsigned)(stride_elems >> 32) & 0xFFFFu); // stride hi16
  g1[7] = 0;
  i32x4 gz; gz[0] = gz[1] = gz[2] = gz[3] = 0;
#if __clang_major__ >= 23
  i32x8 gz8;
#pragma unroll
  for (int i = 0; i < 8; ++i) gz8[i] = 0;
  __builtin_amdgcn_tensor_load_to_lds(g0, g1, gz, gz, gz8, 0);
#else
  __builtin_amdgcn_tensor_load_to_lds(g0, g1, gz, gz, 0);
#endif
}
#endif

// ---------------------------------------------------------------- conversions
__global__ void cvt_f32_bf16(const float* __restrict__ in, bf16* __restrict__ out, int n) {
  int i = blockIdx.x * blockDim.x + threadIdx.x;
  int stride = gridDim.x * blockDim.x;
  for (; i < n; i += stride) out[i] = f2bf(in[i]);
}

// ------------------------------------------------------------------- GEMM
// C[M,N] (f32 rm) = A[M,K] (bf16 rm) * B[K,N] (bf16 rm); N, K compile-time.
// Block tile 128x64, K-step 32, 8 waves; wave = (wm 0..3, wn 0..1) owns a
// 32x32 output patch = 2x2 WMMA tiles. A tiles stream through a double-
// buffered TDM pipeline (DMA of tile t+1 overlaps WMMA on tile t); B tile is
// transposed into [n][k] LDS by the threads.
#define TBM 128
#define TBN 64
#define TBK 32

template <int N, int K>
__global__ __launch_bounds__(256) void gemm_bf16(
    const bf16* __restrict__ A, const bf16* __restrict__ Bm, float* __restrict__ C)
{
  __shared__ __align__(16) bf16 As[2][TBM][TBK];  // 16 KB double buffer
  __shared__ __align__(16) bf16 Bs[TBN][TBK];     //  4 KB, stored [n][k]

  const int bm = blockIdx.y * TBM;
  const int bn = blockIdx.x * TBN;
  const int tid  = threadIdx.x;
  const int wave = tid >> 5, lane = tid & 31;
  const int wm = wave >> 1, wn = wave & 1;
  const int lm = lane & 15, half = lane >> 4;

  v8f acc[2][2];
#pragma unroll
  for (int i = 0; i < 2; ++i)
#pragma unroll
    for (int j = 0; j < 2; ++j) acc[i][j] = v8f_zero();

  constexpr int NT = K / TBK;

#if HAVE_TDM
  if (tid < 32) {                                  // prologue: DMA tile 0
    tdm_load_2d((unsigned)(size_t)(void*)&As[0][0][0],
                (unsigned long long)(size_t)(const void*)(A + (size_t)bm * K),
                TBK, TBM, (unsigned long long)K);
  }
#endif

  for (int t = 0; t < NT; ++t) {
    const int p  = t & 1;
    const int k0 = t * TBK;
#if HAVE_TDM
    if (tid < 32) __builtin_amdgcn_s_wait_tensorcnt(0);  // tile t landed
#endif
    __syncthreads();                               // all waves: tile t visible;
                                                   // buffer p^1 readers done
#if HAVE_TDM
    if (tid < 32 && t + 1 < NT) {                  // DMA tile t+1 (overlapped)
      tdm_load_2d((unsigned)(size_t)(void*)&As[p ^ 1][0][0],
                  (unsigned long long)(size_t)(const void*)(A + (size_t)bm * K + k0 + TBK),
                  TBK, TBM, (unsigned long long)K);
    }
#else
    { // A tile: 128x32, each thread one 16-elem half row
      int r = tid >> 1;
      int c = (tid & 1) << 4;
      const bf16* src = A + (size_t)(bm + r) * K + k0 + c;
      *(bf16x8*)&As[p][r][c]     = *(const bf16x8*)src;
      *(bf16x8*)&As[p][r][c + 8] = *(const bf16x8*)(src + 8);
    }
#endif
    { // B tile: 32(K)x64(N), write transposed into [n][k]
      int kk = tid >> 3;
      int nn = (tid & 7) << 3;
      const bf16* bsrc = Bm + (size_t)(k0 + kk) * N + bn + nn;
      bf16x8 v = *(const bf16x8*)bsrc;
      if (t + 1 < NT) __builtin_prefetch(bsrc + (size_t)TBK * N, 0, 0);
#pragma unroll
      for (int i = 0; i < 8; ++i) Bs[nn + i][kk] = v[i];
    }
    __syncthreads();
    v16bf af[2], bfr[2];
#pragma unroll
    for (int i = 0; i < 2; ++i) af[i]  = frag_a_ld(&As[p][(wm * 2 + i) * 16 + lm][0], half);
#pragma unroll
    for (int j = 0; j < 2; ++j) bfr[j] = frag_b_ld(&Bs[(wn * 2 + j) * 16 + lm][0], half);
#pragma unroll
    for (int i = 0; i < 2; ++i)
#pragma unroll
      for (int j = 0; j < 2; ++j)
        acc[i][j] = __builtin_amdgcn_wmma_f32_16x16x32_bf16(
            false, af[i], false, bfr[j], (short)0, acc[i][j], false, false);
  }
#pragma unroll
  for (int i = 0; i < 2; ++i)
#pragma unroll
    for (int j = 0; j < 2; ++j) {
      const int gm = bm + (wm * 2 + i) * 16 + half * 8;
      const int gn = bn + (wn * 2 + j) * 16 + lm;
      float* cp = C + (size_t)gm * N + gn;      // constant N -> imm offsets
#pragma unroll
      for (int r = 0; r < 8; ++r) cp[(size_t)r * N] = acc[i][j][r];
    }
}

// ------------------------------------------------- fused RMSNorm + RoPE -> bf16
// in: [B, L, NH, HD] f32 ; out: [B, NH, L, HD] bf16
__global__ __launch_bounds__(128) void rmsnorm_rope(
    const float* __restrict__ in, const float* __restrict__ scale,
    const float* __restrict__ cosT, const float* __restrict__ sinT,
    bf16* __restrict__ out, int NH)
{
  __shared__ float sh[HD_];
  const int l = blockIdx.x, hh = blockIdx.y, bz = blockIdx.z;
  const int d = threadIdx.x;
  float v = in[(((size_t)bz * L_ + l) * NH + hh) * HD_ + d];
  sh[d] = v * v;
  __syncthreads();
  for (int s = HD_ / 2; s > 0; s >>= 1) {
    if (d < s) sh[d] += sh[d + s];
    __syncthreads();
  }
  float var = sh[0] * (1.0f / HD_);
  float xn = v * rsqrtf(var + EPS_) * scale[d];
  __syncthreads();
  sh[d] = xn;
  __syncthreads();
  float part = (d < HD_ / 2) ? -sh[d + HD_ / 2] : sh[d - HD_ / 2];
  float o = xn * cosT[(size_t)l * HD_ + d] + part * sinT[(size_t)l * HD_ + d];
  out[(((size_t)bz * NH + hh) * L_ + l) * HD_ + d] = f2bf(o);
}

// V relayout: [B,L,G,HD] f32 -> [B,G,L,HD] bf16
__global__ void v_relayout(const float* __restrict__ in, bf16* __restrict__ out) {
  int i = blockIdx.x * blockDim.x + threadIdx.x;
  const int total = B_ * L_ * G_ * HD_;
  int stride = gridDim.x * blockDim.x;
  for (; i < total; i += stride) {
    int d = i & (HD_ - 1);
    int t = i >> 7;           // /HD_
    int g = t & (G_ - 1);
    int l = (t >> 2) & (L_ - 1);
    int bz = t >> 13;         // /(G_*L_)
    out[(((size_t)bz * G_ + g) * L_ + l) * HD_ + d] = f2bf(in[i]);
  }
}

// ------------------------------------------------------------ flash attention
// Q: [B,H,L,HD] bf16 ; K,V: [B,G,L,HD] bf16 ; O: [B,L,H*HD] bf16
// K tiles stream through a double-buffered TDM pipeline (DMA of block t+1
// overlaps softmax + PV of block t).
#define BQ  64
#define BKV 32

__global__ __launch_bounds__(256) void flash_attn(
    const bf16* __restrict__ Qh, const bf16* __restrict__ Kh,
    const bf16* __restrict__ Vh, bf16* __restrict__ Oh)
{
  __shared__ __align__(16) bf16 Qs[BQ][HD_];       // 16 KB
  __shared__ __align__(16) bf16 Ks[2][BKV][HD_];   // 16 KB double buffer
  __shared__ __align__(16) bf16 Vt[HD_][BKV];      //  8 KB (transposed V)
  __shared__ __align__(16) float Sb[BQ][BKV];      //  8 KB
  __shared__ __align__(16) bf16 Pb[BQ][BKV];       //  4 KB
  __shared__ float rowm[BQ], rowl[BQ], rowfac[BQ];

  const int qb = blockIdx.x, hz = blockIdx.y, bz = blockIdx.z;
  const int g  = hz / (H_ / G_);
  const int q0 = qb * BQ;
  const int tid  = threadIdx.x;
  const int wave = tid >> 5, lane = tid & 31;
  const int wm = wave >> 1, wn = wave & 1;
  const int lm = lane & 15, half = lane >> 4;

  const bf16* Qbase = Qh + ((size_t)bz * H_ + hz) * L_ * HD_;
  const bf16* Kbase = Kh + ((size_t)bz * G_ + g) * L_ * HD_;
  const bf16* Vbase = Vh + ((size_t)bz * G_ + g) * L_ * HD_;

#if HAVE_TDM
  if (tid < 32) {                                  // prologue: Q tile + K block 0
    tdm_load_2d((unsigned)(size_t)(void*)&Qs[0][0],
                (unsigned long long)(size_t)(const void*)(Qbase + (size_t)q0 * HD_),
                HD_, BQ, HD_);
    tdm_load_2d((unsigned)(size_t)(void*)&Ks[0][0][0],
                (unsigned long long)(size_t)(const void*)Kbase,
                HD_, BKV, HD_);
  }
#else
  { // Q tile 64x128
    int r = tid >> 2;
    int c = (tid & 3) << 5;
    const bf16* src = Qbase + (size_t)(q0 + r) * HD_ + c;
#pragma unroll
    for (int i = 0; i < 4; ++i)
      *(bf16x8*)&Qs[r][c + i * 8] = *(const bf16x8*)(src + i * 8);
  }
#endif
  if (tid < BQ) { rowm[tid] = -3.0e38f; rowl[tid] = 0.0f; }

  v8f acc[4];
#pragma unroll
  for (int j = 0; j < 4; ++j) acc[j] = v8f_zero();

  const int nkv = (q0 + BQ) / BKV;                 // causal: skip fully-masked
  for (int t = 0; t < nkv; ++t) {
    const int kv0 = t * BKV;
    const int p = t & 1;
#if HAVE_TDM
    if (tid < 32) __builtin_amdgcn_s_wait_tensorcnt(0);   // K block t (+Q) landed
#endif
    __syncthreads();                               // visible; Pb/Vt/Ks reuse safe
#if HAVE_TDM
    if (tid < 32 && t + 1 < nkv) {                 // DMA K block t+1 (overlapped)
      tdm_load_2d((unsigned)(size_t)(void*)&Ks[p ^ 1][0][0],
                  (unsigned long long)(size_t)(const void*)(Kbase + (size_t)(kv0 + BKV) * HD_),
                  HD_, BKV, HD_);
    }
    { // transposed V tile (all threads)
      int rr = tid >> 3;
      int cc = (tid & 7) << 4;
      const bf16* vsrc = Vbase + (size_t)(kv0 + rr) * HD_ + cc;
      bf16x8 v0 = *(const bf16x8*)vsrc;
      bf16x8 v1 = *(const bf16x8*)(vsrc + 8);
#pragma unroll
      for (int i = 0; i < 8; ++i) { Vt[cc + i][rr] = v0[i]; Vt[cc + 8 + i][rr] = v1[i]; }
    }
#else
    { // K tile + transposed V tile (32 rows x 128)
      int rr = tid >> 3;
      int cc = (tid & 7) << 4;
      const bf16* ksrc = Kbase + (size_t)(kv0 + rr) * HD_ + cc;
      *(bf16x8*)&Ks[p][rr][cc]     = *(const bf16x8*)ksrc;
      *(bf16x8*)&Ks[p][rr][cc + 8] = *(const bf16x8*)(ksrc + 8);
      const bf16* vsrc = Vbase + (size_t)(kv0 + rr) * HD_ + cc;
      bf16x8 v0 = *(const bf16x8*)vsrc;
      bf16x8 v1 = *(const bf16x8*)(vsrc + 8);
#pragma unroll
      for (int i = 0; i < 8; ++i) { Vt[cc + i][rr] = v0[i]; Vt[cc + 8 + i][rr] = v1[i]; }
    }
#endif
    __syncthreads();
    { // S = Q K^T (scaled): wave (wm,wn) -> 16x16 tile (wm,wn) of 64x32 S
      v8f s = v8f_zero();
#pragma unroll
      for (int k = 0; k < HD_; k += 32) {
        v16bf a  = frag_a_ld(&Qs[wm * 16 + lm][k], half);
        v16bf bb = frag_b_ld(&Ks[p][wn * 16 + lm][k], half);  // lane = kv column
        s = __builtin_amdgcn_wmma_f32_16x16x32_bf16(false, a, false, bb, (short)0, s, false, false);
      }
#pragma unroll
      for (int r = 0; r < 8; ++r)
        Sb[wm * 16 + half * 8 + r][wn * 16 + lm] = s[r] * SM_SCALE;
    }
    __syncthreads();
    if (tid < BQ) { // online-softmax row update
      const int row = tid, qi = q0 + row;
      const int cmax = qi - kv0;                   // last valid col (causal)
      float mold = rowm[row];
      float mmax = mold;
      for (int c = 0; c < BKV; ++c)
        if (c <= cmax) mmax = fmaxf(mmax, Sb[row][c]);
      float fac  = __expf(mold - mmax);
      float lsum = rowl[row] * fac;
      for (int c = 0; c < BKV; ++c) {
        float p2 = (c <= cmax) ? __expf(Sb[row][c] - mmax) : 0.0f;
        lsum += p2;
        Pb[row][c] = f2bf(p2);
      }
      rowm[row] = mmax; rowl[row] = lsum; rowfac[row] = fac;
    }
    __syncthreads();
    { // O = O*fac + P V : wave (wm,wn) -> n-tiles wn*4..wn*4+3 of 64x128 O
      float fr[8];
#pragma unroll
      for (int r = 0; r < 8; ++r) fr[r] = rowfac[wm * 16 + half * 8 + r];
      v16bf ap = frag_a_ld(&Pb[wm * 16 + lm][0], half);
#pragma unroll
      for (int j = 0; j < 4; ++j) {
#pragma unroll
        for (int r = 0; r < 8; ++r) acc[j][r] *= fr[r];
        v16bf bv = frag_b_ld(&Vt[(wn * 4 + j) * 16 + lm][0], half);
        acc[j] = __builtin_amdgcn_wmma_f32_16x16x32_bf16(false, ap, false, bv, (short)0, acc[j], false, false);
      }
    }
  }
  __syncthreads();
  { // epilogue: O /= l, write [B, L, H*HD] (constant stride -> imm offsets)
    float il[8];
#pragma unroll
    for (int r = 0; r < 8; ++r) il[r] = 1.0f / rowl[wm * 16 + half * 8 + r];
    const int gq = q0 + wm * 16 + half * 8;
#pragma unroll
    for (int j = 0; j < 4; ++j) {
      const int gd = hz * HD_ + (wn * 4 + j) * 16 + lm;
      bf16* op = Oh + ((size_t)bz * L_ + gq) * (H_ * HD_) + gd;
#pragma unroll
      for (int r = 0; r < 8; ++r)
        op[(size_t)r * (H_ * HD_)] = f2bf(acc[j][r] * il[r]);
    }
  }
}

// ------------------------------------------------------------------ launcher
extern "C" void kernel_launch(void* const* d_in, const int* in_sizes, int n_in,
                              void* d_out, int out_size, void* d_ws, size_t ws_size,
                              hipStream_t stream)
{
  (void)in_sizes; (void)n_in; (void)out_size; (void)ws_size;
  const float* x    = (const float*)d_in[0];
  const float* Wq   = (const float*)d_in[1];
  const float* Wk   = (const float*)d_in[2];
  const float* Wv   = (const float*)d_in[3];
  const float* Wo   = (const float*)d_in[4];
  const float* qs   = (const float*)d_in[5];
  const float* ksc  = (const float*)d_in[6];
  const float* cosT = (const float*)d_in[7];
  const float* sinT = (const float*)d_in[8];
  float* out = (float*)d_out;

  char* ws = (char*)d_ws;
  size_t off = 0;
  auto alloc = [&](size_t bytes) -> char* {
    char* p = ws + off;
    off += (bytes + 255) & ~(size_t)255;
    return p;
  };

  const size_t ML = (size_t)B_ * L_;              // 4096 rows
  bf16*  x_bf   = (bf16*)alloc(ML * D_ * 2);
  bf16*  Wq_bf  = (bf16*)alloc((size_t)D_ * (H_ * HD_) * 2);
  bf16*  Wk_bf  = (bf16*)alloc((size_t)D_ * (G_ * HD_) * 2);
  bf16*  Wv_bf  = (bf16*)alloc((size_t)D_ * (G_ * HD_) * 2);
  bf16*  Wo_bf  = (bf16*)alloc((size_t)(H_ * HD_) * D_ * 2);
  float* q_f    = (float*)alloc(ML * (H_ * HD_) * 4);
  float* k_f    = (float*)alloc(ML * (G_ * HD_) * 4);
  float* v_f    = (float*)alloc(ML * (G_ * HD_) * 4);
  bf16*  q_hat  = (bf16*)alloc(ML * (H_ * HD_) * 2);
  bf16*  k_hat  = (bf16*)alloc(ML * (G_ * HD_) * 2);
  bf16*  v_hat  = (bf16*)alloc(ML * (G_ * HD_) * 2);
  bf16*  attn   = (bf16*)alloc(ML * (H_ * HD_) * 2);

  // 1) f32 -> bf16 staging
  cvt_f32_bf16<<<1024, 256, 0, stream>>>(x,  x_bf,  (int)(ML * D_));
  cvt_f32_bf16<<<1024, 256, 0, stream>>>(Wq, Wq_bf, D_ * H_ * HD_);
  cvt_f32_bf16<<<512,  256, 0, stream>>>(Wk, Wk_bf, D_ * G_ * HD_);
  cvt_f32_bf16<<<512,  256, 0, stream>>>(Wv, Wv_bf, D_ * G_ * HD_);
  cvt_f32_bf16<<<1024, 256, 0, stream>>>(Wo, Wo_bf, H_ * HD_ * D_);

  // 2) projections (WMMA, A-tiles via pipelined TDM)
  gemm_bf16<H_ * HD_, D_><<<dim3((H_ * HD_) / TBN, ML / TBM), 256, 0, stream>>>(
      x_bf, Wq_bf, q_f);
  gemm_bf16<G_ * HD_, D_><<<dim3((G_ * HD_) / TBN, ML / TBM), 256, 0, stream>>>(
      x_bf, Wk_bf, k_f);
  gemm_bf16<G_ * HD_, D_><<<dim3((G_ * HD_) / TBN, ML / TBM), 256, 0, stream>>>(
      x_bf, Wv_bf, v_f);

  // 3) RMSNorm + RoPE -> bf16 (layout [B,NH,L,HD]); V relayout
  rmsnorm_rope<<<dim3(L_, H_, B_), 128, 0, stream>>>(q_f, qs,  cosT, sinT, q_hat, H_);
  rmsnorm_rope<<<dim3(L_, G_, B_), 128, 0, stream>>>(k_f, ksc, cosT, sinT, k_hat, G_);
  v_relayout<<<1024, 256, 0, stream>>>(v_f, v_hat);

  // 4) causal GQA flash attention (WMMA, Q/K tiles via pipelined TDM)
  flash_attn<<<dim3(L_ / BQ, H_, B_), 256, 0, stream>>>(q_hat, k_hat, v_hat, attn);

  // 5) output projection (WMMA) -> f32
  gemm_bf16<D_, H_ * HD_><<<dim3(D_ / TBN, ML / TBM), 256, 0, stream>>>(
      attn, Wo_bf, out);
}
